// SwinBlock_944892805591
// MI455X (gfx1250) — compile-verified, matrix operand order checked by reading, not run on hardware
//
#include <hip/hip_runtime.h>
#include <hip/hip_bf16.h>

// Swin block for MI455X (gfx1250, wave32, WMMA + TDM).
// Pipeline: pack(weights->f16 fragment-major) -> prep(LN1+shift+partition,f16)
//           -> bias(rpb MLP + mask) -> attn(QKV+QK^T+softmax+PV fused, TDM X tile)
//           -> proj(+reverse+residual, TDM X tile) -> mlp(LN2+fc1+gelu+fc2 fused).

typedef __attribute__((ext_vector_type(16))) _Float16 v16h;
typedef __attribute__((ext_vector_type(8)))  _Float16 v8h;
typedef __attribute__((ext_vector_type(8)))  float    v8f;
typedef __attribute__((ext_vector_type(4))) unsigned int u32x4;
typedef __attribute__((ext_vector_type(4))) int          i32x4;
typedef __attribute__((ext_vector_type(8))) int          i32x8;

#define DIM 256
#define HEADS 8
#define HEAD_DIM 32
#define NTOK 64
#define NW 64
#define BATCH 32
#define NWIN_TOT (BATCH*NW)      // 2048
#define TOK_TOT (NWIN_TOT*NTOK)  // 131072
#define HID 1024

#if defined(__has_builtin)
#if __has_builtin(__builtin_amdgcn_tensor_load_to_lds)
#define HAVE_TDM 1
#endif
#endif
#ifndef HAVE_TDM
#define HAVE_TDM 0
#endif

// ---- Tensor Data Mover: 1-D contiguous global -> LDS copy -------------------
// D# per CDNA5 ISA §8: group0 {count=1, lds_addr, global_addr, type=2},
// group1 {data_size=8B, tensor_dim0=tile_dim0=bytes/8, dim1=1, strides}, g2/g3=0.
__device__ __forceinline__ void tdm_copy_to_lds(void* lds_ptr, const void* gptr,
                                                unsigned bytes) {
#if HAVE_TDM
  unsigned lds = (unsigned)(size_t)lds_ptr;       // low 32 bits = LDS offset
  unsigned long long ga = (unsigned long long)(size_t)gptr;
  unsigned n8 = bytes >> 3;                       // 8-byte elements
  u32x4 g0 = {1u, lds, (unsigned)ga,
              (unsigned)((ga >> 32) & 0x01FFFFFFu) | 0x80000000u};
  i32x8 g1;
  g1[0] = (int)(3u << 16);                        // data_size = 8B, no flags
  g1[1] = (int)((n8 & 0xFFFFu) << 16);            // tensor_dim0 lo16 at bit48
  g1[2] = (int)(((n8 >> 16) & 0xFFFFu) | (1u << 16));  // dim0 hi | tensor_dim1=1
  g1[3] = (int)((n8 & 0xFFFFu) << 16);            // tile_dim0
  g1[4] = 1;                                      // tile_dim1=1, tile_dim2=0
  g1[5] = (int)n8;                                // tensor_dim0_stride lo32
  g1[6] = (int)((n8 & 0xFFFFu) << 16);            // tensor_dim1_stride lo16
  g1[7] = 0;
  i32x4 gz = {0, 0, 0, 0};
#if __has_include(<hip/amd_detail/amd_gfx1250_TDM.h>)
  i32x8 g4z = {0, 0, 0, 0, 0, 0, 0, 0};
  __builtin_amdgcn_tensor_load_to_lds(g0, g1, gz, gz, g4z, 0);
#else
  __builtin_amdgcn_tensor_load_to_lds(g0, g1, gz, gz, 0);
#endif
#else
  (void)lds_ptr; (void)gptr; (void)bytes;
#endif
}

// Stage one contiguous tile into LDS: TDM on wave 0 when available, else all
// threads copy with b128s. Always followed by __syncthreads() at the call site.
__device__ __forceinline__ void stage_tile(void* lds_ptr, const void* gptr,
                                           unsigned bytes, int tid, int nthreads) {
#if HAVE_TDM
  if ((tid >> 5) == 0) {
    tdm_copy_to_lds(lds_ptr, gptr, bytes);
    __builtin_amdgcn_s_wait_tensorcnt(0);
  }
#else
  for (unsigned idx = tid; idx < (bytes >> 4); idx += nthreads)
    ((float4*)lds_ptr)[idx] = ((const float4*)gptr)[idx];
#endif
  (void)nthreads;
}

// ---- fragment helpers -------------------------------------------------------
// 16-bit A/B fragment = two contiguous 8-half runs at (k0+8*half) and +16.
__device__ __forceinline__ v16h load_frag16(const _Float16* p) {
  v8h lo = *(const v8h*)p;
  v8h hi = *(const v8h*)(p + 16);
  return __builtin_shufflevector(lo, hi, 0, 1, 2, 3, 4, 5, 6, 7,
                                 8, 9, 10, 11, 12, 13, 14, 15);
}

__device__ __forceinline__ v16h load_a_rm(const _Float16* src, int stride, int k0) {
  int l = threadIdx.x & 31;
  return load_frag16(src + (l & 15) * stride + k0 + (l >> 4) * 8);
}

__device__ __forceinline__ v16h load_b_nk_h(const _Float16* src, int ldk) {
  int l = threadIdx.x & 31;
  return load_frag16(src + (l & 15) * ldk + (l >> 4) * 8);
}

__device__ __forceinline__ v16h load_b_pk(const _Float16* base, int tile) {
  int l = threadIdx.x & 31;
  return *((const v16h*)base + (size_t)tile * 32 + l);
}

__device__ __forceinline__ v8f wmma16(v16h a, v16h b, v8f c) {
  return __builtin_amdgcn_wmma_f32_16x16x32_f16(false, a, false, b, (short)0, c,
                                                false, false);
}

__device__ __forceinline__ float wave_sum(float v) {
#pragma unroll
  for (int m = 16; m >= 1; m >>= 1) v += __shfl_xor(v, m, 32);
  return v;
}

// ---------------- kernel 0: pack f32 NxK weights -> f16 fragment-major -------
__global__ __launch_bounds__(256) void k_pack(const float* __restrict__ src,
                                              _Float16* __restrict__ dst,
                                              int N, int K) {
  int idx = blockIdx.x * 256 + threadIdx.x;
  if (idx >= N * K) return;
  int j = idx & 15;
  int lane = (idx >> 4) & 31;
  int tile = idx >> 9;
  int numKT = K >> 5;
  int kt = tile % numKT, nt = tile / numKT;
  int k = kt * 32 + (lane >> 4) * 8 + ((j < 8) ? j : (8 + j));
  int n = nt * 16 + (lane & 15);
  dst[idx] = (_Float16)src[(size_t)n * K + k];
}

// ---------------- kernel 1: LN1 + cyclic shift + window partition -> f16 -----
__global__ __launch_bounds__(256) void k_prep(const float* __restrict__ x,
                                              const float* __restrict__ n1w,
                                              const float* __restrict__ n1b,
                                              _Float16* __restrict__ xw) {
  int wave = threadIdx.x >> 5, l = threadIdx.x & 31;
  int o = blockIdx.x * 8 + wave;          // output token id [0, 131072)
  int gw = o >> 6, n = o & 63;
  int b = gw >> 6, wi = gw & 63;
  int hs = (wi >> 3) * 8 + (n >> 3);
  int ws = (wi & 7) * 8 + (n & 7);
  int h = (hs + 4) & 63, w = (ws + 4) & 63;   // roll(-SHIFT)
  const float* src = x + ((size_t)b * 4096 + (size_t)h * 64 + w) * DIM;
  float4 f0 = ((const float4*)src)[l * 2];
  float4 f1 = ((const float4*)src)[l * 2 + 1];
  float v[8] = {f0.x, f0.y, f0.z, f0.w, f1.x, f1.y, f1.z, f1.w};
  float s = v[0] + v[1] + v[2] + v[3] + v[4] + v[5] + v[6] + v[7];
  s = wave_sum(s);
  float mu = s * (1.f / 256.f);
  float q = 0.f;
#pragma unroll
  for (int j = 0; j < 8; ++j) { float d = v[j] - mu; q += d * d; }
  q = wave_sum(q);
  float rs = rsqrtf(q * (1.f / 256.f) + 1e-5f);
  float4 w0 = ((const float4*)n1w)[l * 2], w1 = ((const float4*)n1w)[l * 2 + 1];
  float4 c0 = ((const float4*)n1b)[l * 2], c1 = ((const float4*)n1b)[l * 2 + 1];
  float gw_[8] = {w0.x, w0.y, w0.z, w0.w, w1.x, w1.y, w1.z, w1.w};
  float gb_[8] = {c0.x, c0.y, c0.z, c0.w, c1.x, c1.y, c1.z, c1.w};
  v8h hv;
#pragma unroll
  for (int j = 0; j < 8; ++j) hv[j] = (_Float16)((v[j] - mu) * rs * gw_[j] + gb_[j]);
  *(v8h*)(xw + (size_t)o * DIM + l * 8) = hv;
}

// ---------------- kernel 2: relative-position-bias MLP + shift mask ----------
__global__ __launch_bounds__(256) void k_bias(const float* __restrict__ pos,
                                              const float* __restrict__ pm1w,
                                              const float* __restrict__ pm1b,
                                              const float* __restrict__ pm2w,
                                              const float* __restrict__ pm2b,
                                              float* __restrict__ bias) {
  __shared__ float py[64], px[64];
  __shared__ int lbl[64];
  __shared__ float w1[64], b1[32], w2[256], b2[8];
  int tid = threadIdx.x;
  int wi = blockIdx.x;
  if (tid < 64) {
    int hs = (wi >> 3) * 8 + (tid >> 3);
    int ws = (wi & 7) * 8 + (tid & 7);
    int h = (hs + 4) & 63, w = (ws + 4) & 63;
    py[tid] = pos[(h * 64 + w) * 2 + 0];
    px[tid] = pos[(h * 64 + w) * 2 + 1];
    int rh = hs < 56 ? 0 : (hs < 60 ? 1 : 2);
    int rw = ws < 56 ? 0 : (ws < 60 ? 1 : 2);
    lbl[tid] = rh * 3 + rw;
    w1[tid] = pm1w[tid];
  }
  if (tid < 32) b1[tid] = pm1b[tid];
  if (tid >= 64 && tid < 320) w2[tid - 64] = pm2w[tid - 64];
  if (tid < 8) b2[tid] = pm2b[tid];
  __syncthreads();
  for (int pr = tid; pr < 4096; pr += 256) {
    int i = pr >> 6, j = pr & 63;
    float dy = py[j] - py[i], dx = px[j] - px[i];
    dy = copysignf(log1pf(fabsf(dy)), dy);
    dx = copysignf(log1pf(fabsf(dx)), dx);
    float msk = (lbl[i] == lbl[j]) ? 0.f : -100.f;
    float r[8];
#pragma unroll
    for (int hh = 0; hh < 8; ++hh) r[hh] = b2[hh];
#pragma unroll
    for (int o2 = 0; o2 < 32; ++o2) {
      float h1 = fmaxf(w1[o2 * 2] * dy + w1[o2 * 2 + 1] * dx + b1[o2], 0.f);
#pragma unroll
      for (int hh = 0; hh < 8; ++hh) r[hh] += w2[hh * 32 + o2] * h1;
    }
#pragma unroll
    for (int hh = 0; hh < 8; ++hh)
      bias[(((size_t)wi * 8 + hh) * 64 + i) * 64 + j] = r[hh] + msk;
  }
}

// ---------------- kernel 3: fused QKV + attention per (window, head) ---------
__global__ __launch_bounds__(128) void k_attn(const _Float16* __restrict__ xw,
                                              const _Float16* __restrict__ qkvp,
                                              const float* __restrict__ qkvb,
                                              const float* __restrict__ bias,
                                              _Float16* __restrict__ aout) {
  __shared__ _Float16 Qs[64 * 32], Ks[64 * 32];
  __shared__ _Float16 Vt[32 * 64];   // V transposed: [d][token]
  __shared__ float rmax[64], rsum[64];
  __shared__ __align__(16) unsigned char ubuf[64 * 256 * 2];  // X / (S,P) overlay
  _Float16* Xs = (_Float16*)ubuf;
  float* Ss = (float*)ubuf;                       // 16KB after X consumed
  _Float16* Ps = (_Float16*)(ubuf + 64 * 64 * 4); // 8KB

  int gw = blockIdx.x >> 3;
  int head = blockIdx.x & 7;
  int wi = gw & 63;
  int tid = threadIdx.x, wave = tid >> 5, l = tid & 31;
  int half = l >> 4, col = l & 15;

  stage_tile(Xs, xw + (size_t)gw * 64 * DIM, 64 * DIM * 2, tid, 128);
  __syncthreads();

  // q,k,v = x @ Wqkv^T (+bias), q scaled — each wave owns a 16-row stripe
  const float scale = 0.17677669529663687f;  // HEAD_DIM^-0.5
  for (int sel = 0; sel < 3; ++sel) {
    v8f acc0 = {}, acc1 = {};
    int ntb = sel * 16 + head * 2;  // 16-col tile index in packed qkv (numKT=8)
#pragma unroll
    for (int kt = 0; kt < 8; ++kt) {
      v16h a = load_a_rm(Xs + wave * 16 * DIM, DIM, kt * 32);
      v16h b0 = load_b_pk(qkvp, ntb * 8 + kt);
      v16h b1 = load_b_pk(qkvp, (ntb + 1) * 8 + kt);
      acc0 = wmma16(a, b0, acc0);
      acc1 = wmma16(a, b1, acc1);
    }
    const float* bb = qkvb + sel * 256 + head * 32;
#pragma unroll
    for (int r = 0; r < 8; ++r) {
      int m = wave * 16 + r + 8 * half;
      float v0 = acc0[r] + bb[col];
      float v1 = acc1[r] + bb[16 + col];
      if (sel == 0) { v0 *= scale; v1 *= scale; }
      if (sel == 2) {                       // store V transposed [d][token]
        Vt[col * 64 + m] = (_Float16)v0;
        Vt[(16 + col) * 64 + m] = (_Float16)v1;
      } else {
        _Float16* dst = (sel == 0) ? Qs : Ks;
        dst[m * 32 + col] = (_Float16)v0;
        dst[m * 32 + 16 + col] = (_Float16)v1;
      }
    }
  }
  __syncthreads();  // X fully consumed; S reuses its buffer

  // S = q @ k^T + rpb + mask
  const float* bptr = bias + (size_t)(wi * 8 + head) * 4096;
#pragma unroll
  for (int nt = 0; nt < 4; ++nt) {
    v16h a = load_a_rm(Qs + wave * 16 * 32, 32, 0);
    v16h b = load_b_nk_h(Ks + nt * 16 * 32, 32);
    v8f s = {};
    s = wmma16(a, b, s);
#pragma unroll
    for (int r = 0; r < 8; ++r) {
      int m = wave * 16 + r + 8 * half, n = nt * 16 + col;
      Ss[m * 64 + n] = s[r] + bptr[m * 64 + n];
    }
  }
  __syncthreads();

  // softmax rows
  if (tid < 64) {
    float mx = -1e30f;
    for (int j = 0; j < 64; ++j) mx = fmaxf(mx, Ss[tid * 64 + j]);
    float sm = 0.f;
    for (int j = 0; j < 64; ++j) sm += __expf(Ss[tid * 64 + j] - mx);
    rmax[tid] = mx;
    rsum[tid] = 1.f / sm;
  }
  __syncthreads();
  for (int idx = tid; idx < 4096; idx += 128) {
    int i = idx >> 6;
    Ps[idx] = (_Float16)(__expf(Ss[idx] - rmax[i]) * rsum[i]);
  }
  __syncthreads();

  // O = P @ V  (M=64, K=64, N=32); B[k][n] = V[k][n] = Vt[n][k]
  v8f o0 = {}, o1 = {};
#pragma unroll
  for (int kt = 0; kt < 2; ++kt) {
    v16h a = load_a_rm(Ps + wave * 16 * 64, 64, kt * 32);
    v16h b0 = load_b_nk_h(Vt + 0 * 64 + kt * 32, 64);
    v16h b1 = load_b_nk_h(Vt + 16 * 64 + kt * 32, 64);
    o0 = wmma16(a, b0, o0);
    o1 = wmma16(a, b1, o1);
  }
  _Float16* dst = aout + (size_t)gw * 64 * DIM + head * 32;
#pragma unroll
  for (int r = 0; r < 8; ++r) {
    int m = wave * 16 + r + 8 * half;
    dst[m * DIM + col] = (_Float16)o0[r];
    dst[m * DIM + 16 + col] = (_Float16)o1[r];
  }
}

// ---------------- kernel 4: proj GEMM + window reverse + unshift + residual --
__global__ __launch_bounds__(256) void k_proj(const _Float16* __restrict__ aout,
                                              const _Float16* __restrict__ pwp,
                                              const float* __restrict__ pb,
                                              const float* __restrict__ x0,
                                              float* __restrict__ out) {
  __shared__ __align__(16) _Float16 Xs[64 * DIM];
  int gw = blockIdx.x;
  int tid = threadIdx.x, wave = tid >> 5, l = tid & 31;
  int mi = wave >> 1, nh = wave & 1;
  int half = l >> 4, col = l & 15;
  stage_tile(Xs, aout + (size_t)gw * 64 * DIM, 64 * DIM * 2, tid, 256);
  __syncthreads();
  v8f acc[8] = {};
#pragma unroll
  for (int kt = 0; kt < 8; ++kt) {
    v16h a = load_a_rm(Xs + mi * 16 * DIM, DIM, kt * 32);
#pragma unroll
    for (int nt = 0; nt < 8; ++nt) {
      v16h b = load_b_pk(pwp, (nh * 8 + nt) * 8 + kt);  // numKT=8
      acc[nt] = wmma16(a, b, acc[nt]);
    }
  }
  int b = gw >> 6, wi = gw & 63;
#pragma unroll
  for (int r = 0; r < 8; ++r) {
    int m = mi * 16 + r + 8 * half;
    int hs = (wi >> 3) * 8 + (m >> 3), ws = (wi & 7) * 8 + (m & 7);
    int h = (hs + 4) & 63, w = (ws + 4) & 63;  // roll(+SHIFT)
    size_t row = ((size_t)b * 4096 + (size_t)h * 64 + w) * DIM;
#pragma unroll
    for (int nt = 0; nt < 8; ++nt) {
      int c = nh * 128 + nt * 16 + col;
      out[row + c] = x0[row + c] + acc[nt][r] + pb[c];
    }
  }
}

// ---------------- kernel 5: fused LN2 + fc1 + GELU + fc2 + residual ----------
__global__ __launch_bounds__(256) void k_mlp(const float* __restrict__ xio,
                                             const float* __restrict__ n2w,
                                             const float* __restrict__ n2b,
                                             const _Float16* __restrict__ w1p,
                                             const float* __restrict__ b1,
                                             const _Float16* __restrict__ w2p,
                                             const float* __restrict__ b2,
                                             float* __restrict__ out) {
  __shared__ _Float16 Xt[64 * DIM];
  __shared__ _Float16 Hs[64 * 64];
  int tid = threadIdx.x, wave = tid >> 5, l = tid & 31;
  size_t tbase = (size_t)blockIdx.x * 64;
  float4 w0 = ((const float4*)n2w)[l * 2], w1v = ((const float4*)n2w)[l * 2 + 1];
  float4 c0 = ((const float4*)n2b)[l * 2], c1 = ((const float4*)n2b)[l * 2 + 1];
  float gw_[8] = {w0.x, w0.y, w0.z, w0.w, w1v.x, w1v.y, w1v.z, w1v.w};
  float gb_[8] = {c0.x, c0.y, c0.z, c0.w, c1.x, c1.y, c1.z, c1.w};
  for (int s = 0; s < 8; ++s) {
    int m = wave * 8 + s;
    const float* src = xio + (tbase + m) * DIM;
    float4 f0 = ((const float4*)src)[l * 2];
    float4 f1 = ((const float4*)src)[l * 2 + 1];
    float v[8] = {f0.x, f0.y, f0.z, f0.w, f1.x, f1.y, f1.z, f1.w};
    float sum = v[0] + v[1] + v[2] + v[3] + v[4] + v[5] + v[6] + v[7];
    sum = wave_sum(sum);
    float mu = sum * (1.f / 256.f);
    float q = 0.f;
#pragma unroll
    for (int j = 0; j < 8; ++j) { float d = v[j] - mu; q += d * d; }
    q = wave_sum(q);
    float rs = rsqrtf(q * (1.f / 256.f) + 1e-5f);
    v8h hv;
#pragma unroll
    for (int j = 0; j < 8; ++j) hv[j] = (_Float16)((v[j] - mu) * rs * gw_[j] + gb_[j]);
    *(v8h*)(Xt + m * DIM + l * 8) = hv;
  }
  __syncthreads();
  int mi = wave >> 1, nh = wave & 1;
  int half = l >> 4, col = l & 15;
  v8f acc[8] = {};
  for (int ch = 0; ch < 16; ++ch) {   // HID chunks of 64: hidden stays on-chip
    int hb = ch * 64;
    v8f a0 = {}, a1 = {};
#pragma unroll
    for (int kt = 0; kt < 8; ++kt) {
      v16h a = load_a_rm(Xt + mi * 16 * DIM, DIM, kt * 32);
      v16h bb0 = load_b_pk(w1p, (ch * 4 + nh * 2 + 0) * 8 + kt);  // numKT=8
      v16h bb1 = load_b_pk(w1p, (ch * 4 + nh * 2 + 1) * 8 + kt);
      a0 = wmma16(a, bb0, a0);
      a1 = wmma16(a, bb1, a1);
    }
#pragma unroll
    for (int r = 0; r < 8; ++r) {
      int m = mi * 16 + r + 8 * half;
      float u0 = a0[r] + b1[hb + nh * 32 + col];
      float u1 = a1[r] + b1[hb + nh * 32 + 16 + col];
      u0 = 0.5f * u0 * (1.f + erff(u0 * 0.70710678118654752f));
      u1 = 0.5f * u1 * (1.f + erff(u1 * 0.70710678118654752f));
      Hs[m * 64 + nh * 32 + col] = (_Float16)u0;
      Hs[m * 64 + nh * 32 + 16 + col] = (_Float16)u1;
    }
    __syncthreads();
#pragma unroll
    for (int kt = 0; kt < 2; ++kt) {
      v16h a = load_a_rm(Hs + mi * 16 * 64, 64, kt * 32);
#pragma unroll
      for (int nt = 0; nt < 8; ++nt) {
        v16h b = load_b_pk(w2p, (nh * 8 + nt) * 32 + (ch * 2 + kt));  // numKT=32
        acc[nt] = wmma16(a, b, acc[nt]);
      }
    }
    __syncthreads();
  }
#pragma unroll
  for (int r = 0; r < 8; ++r) {
    int m = mi * 16 + r + 8 * half;
    size_t row = (tbase + m) * DIM;
#pragma unroll
    for (int nt = 0; nt < 8; ++nt) {
      int c = nh * 128 + nt * 16 + col;
      out[row + c] = xio[row + c] + acc[nt][r] + b2[c];
    }
  }
}

extern "C" void kernel_launch(void* const* d_in, const int* in_sizes, int n_in,
                              void* d_out, int out_size, void* d_ws, size_t ws_size,
                              hipStream_t stream) {
  (void)in_sizes; (void)n_in; (void)out_size; (void)ws_size;
  const float* x     = (const float*)d_in[0];
  const float* pos   = (const float*)d_in[1];
  const float* n1w   = (const float*)d_in[2];
  const float* n1b   = (const float*)d_in[3];
  const float* qkvw  = (const float*)d_in[4];
  const float* qkvb  = (const float*)d_in[5];
  const float* projw = (const float*)d_in[6];
  const float* projb = (const float*)d_in[7];
  const float* pm1w  = (const float*)d_in[8];
  const float* pm1b  = (const float*)d_in[9];
  const float* pm2w  = (const float*)d_in[10];
  const float* pm2b  = (const float*)d_in[11];
  const float* n2w   = (const float*)d_in[12];
  const float* n2b   = (const float*)d_in[13];
  const float* fc1w  = (const float*)d_in[14];
  const float* fc1b  = (const float*)d_in[15];
  const float* fc2w  = (const float*)d_in[16];
  const float* fc2b  = (const float*)d_in[17];
  float* out = (float*)d_out;

  // workspace: xw f16 (64MiB) | aout f16 (64MiB) | bias f32 (8MiB) | packed wts
  char* ws = (char*)d_ws;
  _Float16* xw    = (_Float16*)ws;
  _Float16* aout  = (_Float16*)(ws + (size_t)64 * 1024 * 1024);
  float*    bias  = (float*)(ws + (size_t)128 * 1024 * 1024);
  char* pk        = ws + (size_t)136 * 1024 * 1024;
  _Float16* qkvp  = (_Float16*)(pk);                         // 768x256  = 384KB
  _Float16* projp = (_Float16*)(pk + (size_t)384 * 1024);    // 256x256  = 128KB
  _Float16* fc1p  = (_Float16*)(pk + (size_t)512 * 1024);    // 1024x256 = 512KB
  _Float16* fc2p  = (_Float16*)(pk + (size_t)1024 * 1024);   // 256x1024 = 512KB

  k_pack<<<768 * 256 / 256, 256, 0, stream>>>(qkvw, qkvp, 768, 256);
  k_pack<<<256 * 256 / 256, 256, 0, stream>>>(projw, projp, 256, 256);
  k_pack<<<1024 * 256 / 256, 256, 0, stream>>>(fc1w, fc1p, 1024, 256);
  k_pack<<<256 * 1024 / 256, 256, 0, stream>>>(fc2w, fc2p, 256, 1024);

  k_prep<<<TOK_TOT / 8, 256, 0, stream>>>(x, n1w, n1b, xw);
  k_bias<<<NW, 256, 0, stream>>>(pos, pm1w, pm1b, pm2w, pm2b, bias);
  k_attn<<<NWIN_TOT * HEADS, 128, 0, stream>>>(xw, qkvp, qkvb, bias, aout);
  k_proj<<<NWIN_TOT, 256, 0, stream>>>(aout, projp, projb, x, out);
  k_mlp<<<TOK_TOT / 64, 256, 0, stream>>>(out, n2w, n2b, fc1p, fc1b, fc2p, fc2b, out);
}
